// DynamicSpatialAttention_3015067042094
// MI455X (gfx1250) — matrix operand order repaired and checked by reading, not compiled, not run on hardware
//
#include <hip/hip_runtime.h>
#include <hip/hip_bf16.h>
#include <math.h>

typedef __attribute__((ext_vector_type(16))) _Float16 v16h;
typedef __attribute__((ext_vector_type(8)))  float    v8f;
typedef __attribute__((ext_vector_type(4)))  float    v4f;

#define B_    32
#define C_    256
#define H_    128
#define W_    128
#define HW    (H_ * W_)
#define TILES 16            // reduction tiles per image
#define TPIX  (HW / TILES)  // 1024 pixels per tile = 256 threads x float4

// ---------------------------------------------------------------------------
// Kernel 1: single pass over x producing BOTH reductions.
//   - per-pixel channel sum -> x_mean (scaled by 1/C)
//   - per-channel per-tile spatial partial sums -> partial[B][C][TILES]
// Deterministic (fixed-order tree reduction, no float atomics).
// All x traffic is non-temporal global_load_b128 (512B per wave per issue).
// ---------------------------------------------------------------------------
__global__ __launch_bounds__(256) void k_reduce(const float* __restrict__ x,
                                                float* __restrict__ partial,
                                                float* __restrict__ xmean) {
    const int b    = blockIdx.y;
    const int tile = blockIdx.x;
    const int tid  = threadIdx.x;
    const int lane = tid & 31;
    const int wave = tid >> 5;
    const int base = tile * TPIX;      // pixel offset of this tile

    __shared__ float red[8];
    v4f acc = {0.f, 0.f, 0.f, 0.f};    // per-pixel channel accumulators (4 px/thread)

    for (int c = 0; c < C_; ++c) {
        const v4f* p = (const v4f*)(x + ((size_t)(b * C_ + c) * HW) + base) + tid;
        v4f v = __builtin_nontemporal_load(p);
        acc += v;
        // block-wide spatial sum of this channel's 1024 pixels (wave32 shuffles)
        float s = v.x + v.y + v.z + v.w;
        #pragma unroll
        for (int off = 16; off > 0; off >>= 1) s += __shfl_xor(s, off, 32);
        if (lane == 0) red[wave] = s;
        __syncthreads();
        if (tid == 0) {
            float t = red[0] + red[1] + red[2] + red[3] +
                      red[4] + red[5] + red[6] + red[7];
            partial[((size_t)(b * C_ + c) * TILES) + tile] = t;
        }
        __syncthreads();
    }
    v4f m = acc * (1.0f / C_);
    __builtin_nontemporal_store(m, (v4f*)(xmean + (size_t)b * HW + base) + tid);
}

// ---------------------------------------------------------------------------
// Kernel 2: finalize pooled[B,C] = sum(partials)/HW  (deterministic order)
// ---------------------------------------------------------------------------
__global__ __launch_bounds__(256) void k_pool_finalize(const float* __restrict__ partial,
                                                       float* __restrict__ pooled) {
    int idx = blockIdx.x * 256 + threadIdx.x;
    if (idx < B_ * C_) {
        float s = 0.f;
        #pragma unroll
        for (int t = 0; t < TILES; ++t) s += partial[(size_t)idx * TILES + t];
        pooled[idx] = s * (1.0f / HW);
    }
}

// ---------------------------------------------------------------------------
// WMMA fragment loaders (layouts per CDNA5 ISA 7.12.2, wave32)
// A: 16x32 f16. lane[0..15]=row M, halves 0..7 -> K=kb+h, halves 8..15 -> K=kb+8+h
//    where kb = 0 (lanes 0-15) or 8 (lanes 16-31).
// B: 32x16 f16. lane%16 = column N, same K order as A (kb = 0 or 8 by lane half).
// C/D f32: VGPR r -> row (r or r+8 by lane half), col = lane%16.
// ---------------------------------------------------------------------------
__device__ inline v16h load_a_frag_lds(const _Float16* __restrict__ src, int stride,
                                       int mbase, int k0, int lane) {
    const int m  = mbase + (lane & 15);
    const int kb = (lane < 16) ? 0 : 8;
    v16h a;
    #pragma unroll
    for (int h = 0; h < 16; ++h) {
        int k = k0 + kb + h + ((h >> 3) << 3);   // h<8: +h ; h>=8: +h+8
        a[h] = src[m * stride + k];
    }
    return a;
}

__device__ inline v16h load_b_frag_f32(const float* __restrict__ src, int ncols,
                                       int k0, int nbase, int lane) {
    const int n  = nbase + (lane & 15);
    const int kb = (lane < 16) ? 0 : 16;
    v16h bf;
    #pragma unroll
    for (int h = 0; h < 16; ++h) {
        int k = k0 + (kb >> 1) + h + ((h >> 3) << 3);
        bf[h] = (n < ncols) ? (_Float16)src[(size_t)k * ncols + n] : (_Float16)0.0f;
    }
    return bf;
}

// ---------------------------------------------------------------------------
// Kernel 3: kernel generator, one workgroup, 32 waves.
//   GEMM1: z = relu(pooled @ w1 + b1)   [32,256]x[256,256], f16 WMMA f32-acc
//   GEMM2: kernels = z @ w2 + b2        [32,256]x[256,9] (N padded to 16)
// ---------------------------------------------------------------------------
__global__ __launch_bounds__(1024) void k_genkernels(const float* __restrict__ pooled,
                                                     const float* __restrict__ w1,
                                                     const float* __restrict__ b1,
                                                     const float* __restrict__ w2,
                                                     const float* __restrict__ b2,
                                                     float* __restrict__ kout) {
    __shared__ _Float16 sP[B_ * C_];   // pooled as f16 (A operand)
    __shared__ _Float16 sZ[B_ * C_];   // z as f16 (A operand of GEMM2)

    const int tid = threadIdx.x;
    for (int i = tid; i < B_ * C_; i += 1024) sP[i] = (_Float16)pooled[i];
    __syncthreads();

    const int wave = tid >> 5, lane = tid & 31;
    const int mi = wave >> 4, ni = wave & 15;   // 2 x 16 output tiles

    v8f acc = {};
    #pragma unroll
    for (int ks = 0; ks < 8; ++ks) {
        v16h a  = load_a_frag_lds(sP, C_, mi * 16, ks * 32, lane);
        v16h bm = load_b_frag_f32(w1, C_, ks * 32, ni * 16, lane);
        acc = __builtin_amdgcn_wmma_f32_16x16x32_f16(false, a, false, bm,
                                                     (short)0, acc, false, false);
    }
    {   // epilogue: bias + relu, store z tile as f16
        const int n     = ni * 16 + (lane & 15);
        const int rbase = mi * 16 + ((lane < 16) ? 0 : 8);
        const float bias = b1[n];
        #pragma unroll
        for (int r = 0; r < 8; ++r) {
            float v = acc[r] + bias;
            v = v > 0.f ? v : 0.f;
            sZ[(rbase + r) * C_ + n] = (_Float16)v;
        }
    }
    __syncthreads();

    if (wave < 2) {   // uniform per-wave branch: EXEC all-ones inside
        v8f acc2 = {};
        #pragma unroll
        for (int ks = 0; ks < 8; ++ks) {
            v16h a  = load_a_frag_lds(sZ, C_, wave * 16, ks * 32, lane);
            v16h bm = load_b_frag_f32(w2, 9, ks * 32, 0, lane);   // cols 9..15 zero
            acc2 = __builtin_amdgcn_wmma_f32_16x16x32_f16(false, a, false, bm,
                                                          (short)0, acc2, false, false);
        }
        const int n     = lane & 15;
        const int rbase = wave * 16 + ((lane < 16) ? 0 : 8);
        if (n < 9) {
            const float bias = b2[n];
            #pragma unroll
            for (int r = 0; r < 8; ++r)
                kout[(rbase + r) * 9 + n] = acc2[r] + bias;
        }
    }
}

// ---------------------------------------------------------------------------
// Kernel 4: 3x3 per-sample conv on x_mean -> sigmoid -> out = x * att.
// Tile = 8 full rows (1024 px); each thread owns 4 consecutive pixels across
// all 256 channels: every c-iteration is a coalesced 4KB b128 load + b128
// store slab per block, all non-temporal (1GB of traffic, zero reuse).
// ---------------------------------------------------------------------------
__global__ __launch_bounds__(256) void k_apply(const float* __restrict__ x,
                                               const float* __restrict__ xmean,
                                               const float* __restrict__ kern,
                                               float* __restrict__ out) {
    const int b    = blockIdx.y;
    const int tile = blockIdx.x;          // 16 tiles of 8 rows per image
    const int tid  = threadIdx.x;

    __shared__ float sk[9];
    if (tid < 9) sk[tid] = kern[b * 9 + tid];
    __syncthreads();

    const int p  = tid * 4;               // pixel offset within tile
    const int h  = tile * 8 + (p >> 7);   // 32 threads per 128-wide row
    const int wb = p & 127;               // starting column (multiple of 4)
    const float* xm = xmean + (size_t)b * HW;

    v4f attv;
    #pragma unroll
    for (int j = 0; j < 4; ++j) {
        const int w = wb + j;
        float a = 0.f;
        #pragma unroll
        for (int dy = -1; dy <= 1; ++dy) {
            const int hh = h + dy;
            if (hh < 0 || hh >= H_) continue;
            #pragma unroll
            for (int dx = -1; dx <= 1; ++dx) {
                const int ww = w + dx;
                if (ww < 0 || ww >= W_) continue;
                a += xm[hh * W_ + ww] * sk[(dy + 1) * 3 + (dx + 1)];
            }
        }
        attv[j] = 1.0f / (1.0f + __expf(-a));
    }

    const size_t pix = (size_t)tile * 1024 + p;        // == h*W_ + wb
    const v4f* xp = (const v4f*)(x   + (size_t)b * C_ * HW + pix);
    v4f*       op = (v4f*)      (out + (size_t)b * C_ * HW + pix);
    #pragma unroll 4
    for (int c = 0; c < C_; ++c) {
        v4f v = __builtin_nontemporal_load(xp + (size_t)c * (HW / 4));
        __builtin_nontemporal_store(v * attv, op + (size_t)c * (HW / 4));
    }
}

// ---------------------------------------------------------------------------
extern "C" void kernel_launch(void* const* d_in, const int* in_sizes, int n_in,
                              void* d_out, int out_size, void* d_ws, size_t ws_size,
                              hipStream_t stream) {
    const float* x  = (const float*)d_in[0];   // [32,256,128,128]
    const float* w1 = (const float*)d_in[1];   // [256,256]
    const float* b1 = (const float*)d_in[2];   // [256]
    const float* w2 = (const float*)d_in[3];   // [256,9]
    const float* b2 = (const float*)d_in[4];   // [9]
    float* out = (float*)d_out;

    // workspace layout (floats):
    float* ws       = (float*)d_ws;
    float* partial  = ws;                                   // B*C*TILES = 131072
    float* pooled   = ws + (size_t)B_ * C_ * TILES;         // 8192
    float* xmean    = pooled + (size_t)B_ * C_;             // B*HW = 524288
    float* kern     = xmean + (size_t)B_ * HW;              // B*9 = 288

    k_reduce<<<dim3(TILES, B_), 256, 0, stream>>>(x, partial, xmean);
    k_pool_finalize<<<dim3((B_ * C_ + 255) / 256), 256, 0, stream>>>(partial, pooled);
    k_genkernels<<<1, 1024, 0, stream>>>(pooled, w1, b1, w2, b2, kern);
    k_apply<<<dim3(HW / 1024, B_), 256, 0, stream>>>(x, xmean, kern, out);
}
